// SwinTransformerBlock_3788161155756
// MI455X (gfx1250) — matrix operand order repaired and checked by reading, not compiled, not run on hardware
//
#include <hip/hip_runtime.h>
#include <hip/hip_bf16.h>

typedef __attribute__((ext_vector_type(16))) _Float16 v16h;
typedef __attribute__((ext_vector_type(8)))  _Float16 v8h;
typedef __attribute__((ext_vector_type(8)))  float    v8f;

// ---------------------------------------------------------------------------
// Problem constants
// ---------------------------------------------------------------------------
#define DIMC   256
#define HEADS  8
#define HEADD  32
#define HRES   64
#define WRES   64
#define WS     8
#define SS     4
#define NWIN   64            // windows per batch image
#define NTOK   64            // tokens per window
#define BATCH  8
#define NWTOT  (BATCH*NWIN)  // 512 windows total
#define MROWS  (NWTOT*NTOK)  // 32768 rows in all big GEMMs
#define EPS_RMS 1.1920929e-07f
#define EPS_LN  1e-5f

// ---------------------------------------------------------------------------
// Workspace layout (bytes). Aliased in dependency order; total ~146 MB < L2.
// ---------------------------------------------------------------------------
#define OFF_QKVW   ((size_t)0)            // 768*256*2   = 393216
#define OFF_PROJW  ((size_t)393216)       // 256*256*2   = 131072
#define OFF_FC1W   ((size_t)524288)       // 1024*256*2  = 524288
#define OFF_FC2W   ((size_t)1048576)      // 256*1024*2  = 524288
#define OFF_TAB    ((size_t)1572864)      // 225*8*4     = 7200 (pad)
#define OFF_BIAS   ((size_t)1581056)      // 8*64*64*4   = 131072
#define OFF_XW     ((size_t)1712128)      // 32768*256*2 = 16777216 (later x1_f16)
#define OFF_Q      ((size_t)18489344)     // 4096*64*32*2 = 16777216
#define OFF_K      ((size_t)35266560)     // 16777216
#define OFF_VT     ((size_t)52043776)     // 16777216
#define OFF_Y16    ((size_t)68820992)     // 32768*256*2 = 16777216
#define OFF_YPROJ  OFF_Q                  // 32768*256*4 = 33554432 (aliases q,k)
#define OFF_H16    OFF_Q                  // 32768*1024*2 = 67108864 (aliases q,k,vT,y16)
#define OFF_X1F32  ((size_t)85598208)     // 33554432
#define OFF_X1F16  OFF_XW                 // aliases xw (dead after qkv gemm)
#define OFF_G      ((size_t)119152640)    // 33554432  -> total 152707072

// ---------------------------------------------------------------------------
// WMMA helpers
// ---------------------------------------------------------------------------
__device__ __forceinline__ v8f wmma_f16(v16h a, v16h b, v8f c) {
  return __builtin_amdgcn_wmma_f32_16x16x32_f16(false, a, false, b, (short)0, c,
                                                false, false);
}

// A-fragment: 16x32 tile of row-major f16 (stride in halves).
// lane l<16: row l, K 0..15; lane l>=16: row l-16, K 16..31 (contiguous 32B)
__device__ __forceinline__ v16h load_a_frag(const _Float16* base, int stride, int lane) {
  int row = lane & 15;
  int hi  = (lane >> 4) << 4;                    // 0 or 16
  const _Float16* p = base + (size_t)row * stride + hi;
  return *(const v16h*)(p);
}

// B-fragment for X @ W^T: W row-major [N][K]; column n of B == row n of W.
// lane l<16: col l, K 0..15; lane l>=16: col l-16, K 16..31 (contiguous 32B)
__device__ __forceinline__ v16h load_b_frag(const _Float16* base, int stride, int lane) {
  int col = lane & 15;
  int ko  = (lane >> 4) << 4;                    // 0 or 16
  return *(const v16h*)(base + (size_t)col * stride + ko);
}

__device__ __forceinline__ int region_label(int g) {
  return g < (HRES - WS) ? 0 : (g < (HRES - SS) ? 1 : 2);
}

// ---------------------------------------------------------------------------
// fp32 -> fp16 weight conversion
// ---------------------------------------------------------------------------
__global__ void cvt_f16_kernel(const float* __restrict__ s, _Float16* __restrict__ d, int n) {
  int i = blockIdx.x * 256 + threadIdx.x;
  if (i < n) d[i] = (_Float16)s[i];
}

// ---------------------------------------------------------------------------
// CPB MLP: tab[225][8], then bias[8][64][64] = 16*sigmoid(tab[relidx][h])
// ---------------------------------------------------------------------------
__device__ __forceinline__ float cpb_coord(int idx) {
  float t = (float)(idx - (WS - 1)) * (8.0f / (float)(WS - 1));
  float a = log2f(fabsf(t) + 1.0f) * (1.0f / 3.0f);   // log2(8)=3
  return t > 0.0f ? a : (t < 0.0f ? -a : 0.0f);
}

__global__ void cpb_tab_kernel(const float* __restrict__ w1, const float* __restrict__ b1,
                               const float* __restrict__ w2, float* __restrict__ tab) {
  __shared__ float hid[512];
  int p = blockIdx.x;                 // 0..224
  float ca = cpb_coord(p / 15);
  float cb = cpb_coord(p % 15);
  for (int j = threadIdx.x; j < 512; j += 256) {
    float h = w1[j * 3 + 0] * ca + w1[j * 3 + 1] * cb + b1[j];
    hid[j] = h > 0.0f ? h : 0.0f;
  }
  __syncthreads();
  int head = threadIdx.x >> 5, lane = threadIdx.x & 31;
  float s = 0.0f;
  for (int j = lane; j < 512; j += 32) s += w2[head * 512 + j] * hid[j];
#pragma unroll
  for (int m = 16; m >= 1; m >>= 1) s += __shfl_xor(s, m, 32);
  if (lane == 0) tab[p * 8 + head] = s;
}

__global__ void cpb_bias_kernel(const float* __restrict__ tab, float* __restrict__ bias) {
  int idx = blockIdx.x * 256 + threadIdx.x;          // h*4096 + i*64 + j
  int h = idx >> 12, i = (idx >> 6) & 63, j = idx & 63;
  int dr = (i >> 3) - (j >> 3) + (WS - 1);
  int dc = (i & 7) - (j & 7) + (WS - 1);
  float t = tab[(dr * 15 + dc) * 8 + h];
  bias[idx] = 16.0f / (1.0f + expf(-t));
}

// ---------------------------------------------------------------------------
// Shifted window partition + f16 convert: xw[(w*64+t)*256 + c]
// ---------------------------------------------------------------------------
__global__ void winpart_kernel(const float* __restrict__ x, _Float16* __restrict__ xw) {
  int row = blockIdx.x;                              // 0..32767
  int c = threadIdx.x;
  int w = row >> 6, t = row & 63;
  int b = w >> 6, wl = w & 63;
  int wh = wl >> 3, wc = wl & 7;
  int gr = ((wh << 3) + (t >> 3) + SS) & 63;
  int gc = ((wc << 3) + (t & 7) + SS) & 63;
  xw[(size_t)row * DIMC + c] =
      (_Float16)x[((size_t)(b << 12) + (gr << 6) + gc) * DIMC + c];
}

// ---------------------------------------------------------------------------
// QKV GEMM (32768 x 768 x 256) + bias + per-head RMSNorm on q,k (in-register),
// writes q/k as [wh][64][32] f16 and v transposed as [wh][32][64] f16.
// One wave computes a 32x32 tile; N-tiles align with heads (32 cols/head).
// ---------------------------------------------------------------------------
__global__ void qkv_gemm_kernel(const _Float16* __restrict__ A, const _Float16* __restrict__ W,
                                const float* __restrict__ bias,
                                const float* __restrict__ nqw, const float* __restrict__ nkw,
                                _Float16* __restrict__ qbuf, _Float16* __restrict__ kbuf,
                                _Float16* __restrict__ vtbuf) {
  int lane = threadIdx.x & 31;
  int tile = blockIdx.x * 8 + (threadIdx.x >> 5);
  const int tilesN = (3 * DIMC) / 32;                // 24
  int tm = (tile / tilesN) * 32;
  int tn = (tile % tilesN) * 32;

  v8f acc[2][2] = {};
  for (int k0 = 0; k0 < DIMC; k0 += 32) {
    v16h a0 = load_a_frag(A + (size_t)tm * DIMC + k0, DIMC, lane);
    v16h a1 = load_a_frag(A + (size_t)(tm + 16) * DIMC + k0, DIMC, lane);
    v16h b0 = load_b_frag(W + (size_t)tn * DIMC + k0, DIMC, lane);
    v16h b1 = load_b_frag(W + (size_t)(tn + 16) * DIMC + k0, DIMC, lane);
    acc[0][0] = wmma_f16(a0, b0, acc[0][0]);
    acc[0][1] = wmma_f16(a0, b1, acc[0][1]);
    acc[1][0] = wmma_f16(a1, b0, acc[1][0]);
    acc[1][1] = wmma_f16(a1, b1, acc[1][1]);
  }

  int nlo = lane & 15, hi8 = (lane >> 4) << 3;
  float bv0 = bias[tn + nlo], bv1 = bias[tn + 16 + nlo];
#pragma unroll
  for (int s = 0; s < 2; ++s)
#pragma unroll
    for (int v = 0; v < 8; ++v) { acc[s][0][v] += bv0; acc[s][1][v] += bv1; }

  int type = tn >> 8;                                // 0=q 1=k 2=v
  int head = (tn >> 5) & 7;
  int w = tm >> 6;
  int tb = tm & 63;
  size_t hb = (size_t)(w * HEADS + head);

  if (type < 2) {
    const float* nw = (type == 0) ? nqw : nkw;
    float nw0 = nw[nlo], nw1 = nw[nlo + 16];
    _Float16* dst = ((type == 0) ? qbuf : kbuf) + hb * NTOK * HEADD;
#pragma unroll
    for (int s = 0; s < 2; ++s) {
#pragma unroll
      for (int v = 0; v < 8; ++v) {
        float ss = acc[s][0][v] * acc[s][0][v] + acc[s][1][v] * acc[s][1][v];
#pragma unroll
        for (int m2 = 8; m2 >= 1; m2 >>= 1) ss += __shfl_xor(ss, m2, 32);
        float sc = rsqrtf(ss * (1.0f / (float)HEADD) + EPS_RMS);
        int t = tb + s * 16 + hi8 + v;
        dst[(size_t)t * HEADD + nlo]      = (_Float16)(acc[s][0][v] * sc * nw0);
        dst[(size_t)t * HEADD + nlo + 16] = (_Float16)(acc[s][1][v] * sc * nw1);
      }
    }
  } else {
    _Float16* dst = vtbuf + hb * HEADD * NTOK;       // [d][t]
#pragma unroll
    for (int s = 0; s < 2; ++s)
#pragma unroll
      for (int ns = 0; ns < 2; ++ns) {
        int d = ns * 16 + nlo;
        v8h pv;
#pragma unroll
        for (int v = 0; v < 8; ++v) pv[v] = (_Float16)acc[s][ns][v];
        *(v8h*)(dst + (size_t)d * NTOK + tb + s * 16 + hi8) = pv;
      }
  }
}

// ---------------------------------------------------------------------------
// Attention: one workgroup (4 waves) per (window, head).
// S = q k^T (+bias+mask) -> softmax -> P (via LDS) -> Y = P v -> y_f16 rows.
// ---------------------------------------------------------------------------
__global__ void attn_kernel(const _Float16* __restrict__ qbuf, const _Float16* __restrict__ kbuf,
                            const _Float16* __restrict__ vtbuf, const float* __restrict__ biasbuf,
                            _Float16* __restrict__ ybuf) {
  __shared__ _Float16 P[NTOK * NTOK];                // 8 KB
  int wh = blockIdx.x;                               // window*8 + head
  int w = wh >> 3, h = wh & 7;
  int lane = threadIdx.x & 31, mt = threadIdx.x >> 5;
  int nlo = lane & 15, hi8 = (lane >> 4) << 3;

  const _Float16* q = qbuf + (size_t)wh * NTOK * HEADD;
  const _Float16* k = kbuf + (size_t)wh * NTOK * HEADD;

  // scores: 16-row strip, 4 column tiles, K=32 (one WMMA each)
  v16h a = load_a_frag(q + (size_t)(mt * 16) * HEADD, HEADD, lane);
  v8f s[4];
#pragma unroll
  for (int nt = 0; nt < 4; ++nt) {
    v16h b = load_b_frag(k + (size_t)(nt * 16) * HEADD, HEADD, lane);
    v8f z = {};
    s[nt] = wmma_f16(a, b, z);
  }

  // bias + shift mask (labels in shifted coords)
  int wl = w & 63;
  int whr = wl >> 3, wwc = wl & 7;
#pragma unroll
  for (int nt = 0; nt < 4; ++nt) {
    int j = nt * 16 + nlo;
    int lj = region_label(whr * 8 + (j >> 3)) * 3 + region_label(wwc * 8 + (j & 7));
#pragma unroll
    for (int v = 0; v < 8; ++v) {
      int i = mt * 16 + hi8 + v;
      int li = region_label(whr * 8 + (i >> 3)) * 3 + region_label(wwc * 8 + (i & 7));
      float add = biasbuf[(size_t)((h << 6) + i) * 64 + j];
      if (li != lj) add -= 100.0f;
      s[nt][v] += add;
    }
  }

  // softmax per row (row = 16 lanes x 4 regs); write P to LDS in f16
#pragma unroll
  for (int v = 0; v < 8; ++v) {
    float mx = fmaxf(fmaxf(s[0][v], s[1][v]), fmaxf(s[2][v], s[3][v]));
#pragma unroll
    for (int m2 = 8; m2 >= 1; m2 >>= 1) mx = fmaxf(mx, __shfl_xor(mx, m2, 32));
    float sum = 0.0f;
#pragma unroll
    for (int nt = 0; nt < 4; ++nt) { s[nt][v] = expf(s[nt][v] - mx); sum += s[nt][v]; }
#pragma unroll
    for (int m2 = 8; m2 >= 1; m2 >>= 1) sum += __shfl_xor(sum, m2, 32);
    float inv = 1.0f / sum;
    int i = mt * 16 + hi8 + v;
#pragma unroll
    for (int nt = 0; nt < 4; ++nt)
      P[i * NTOK + nt * 16 + nlo] = (_Float16)(s[nt][v] * inv);
  }
  __syncthreads();

  // Y strip = P_strip (16x64) @ v (64x32): 2 N-tiles x 2 K-steps
  const _Float16* vt = vtbuf + (size_t)wh * HEADD * NTOK;
  v8f y[2] = {};
  for (int k0 = 0; k0 < NTOK; k0 += 32) {
    v16h pa = load_a_frag(&P[mt * 16 * NTOK + k0], NTOK, lane);
#pragma unroll
    for (int nt = 0; nt < 2; ++nt) {
      v16h pb = load_b_frag(vt + (size_t)(nt * 16) * NTOK + k0, NTOK, lane);
      y[nt] = wmma_f16(pa, pb, y[nt]);
    }
  }
#pragma unroll
  for (int nt = 0; nt < 2; ++nt)
#pragma unroll
    for (int v = 0; v < 8; ++v) {
      int m = mt * 16 + hi8 + v;
      int d = nt * 16 + nlo;
      ybuf[(size_t)(w * NTOK + m) * DIMC + h * HEADD + d] = (_Float16)y[nt][v];
    }
}

// ---------------------------------------------------------------------------
// Generic WMMA GEMM: C = A(MxK) @ W^T(KxN) + bias, optional exact GELU,
// f32 and/or f16 output. One wave = 32x32 output tile.
// ---------------------------------------------------------------------------
__global__ void gemm_wmma_kernel(const _Float16* __restrict__ A, const _Float16* __restrict__ W,
                                 const float* __restrict__ bias,
                                 float* __restrict__ outF, _Float16* __restrict__ outH,
                                 int N, int K, int act) {
  int lane = threadIdx.x & 31;
  int tile = blockIdx.x * 8 + (threadIdx.x >> 5);
  int tilesN = N >> 5;
  int tm = (tile / tilesN) << 5;
  int tn = (tile % tilesN) << 5;

  v8f acc[2][2] = {};
  for (int k0 = 0; k0 < K; k0 += 32) {
    v16h a0 = load_a_frag(A + (size_t)tm * K + k0, K, lane);
    v16h a1 = load_a_frag(A + (size_t)(tm + 16) * K + k0, K, lane);
    v16h b0 = load_b_frag(W + (size_t)tn * K + k0, K, lane);
    v16h b1 = load_b_frag(W + (size_t)(tn + 16) * K + k0, K, lane);
    acc[0][0] = wmma_f16(a0, b0, acc[0][0]);
    acc[0][1] = wmma_f16(a0, b1, acc[0][1]);
    acc[1][0] = wmma_f16(a1, b0, acc[1][0]);
    acc[1][1] = wmma_f16(a1, b1, acc[1][1]);
  }

  int nlo = lane & 15, hi8 = (lane >> 4) << 3;
  float bv[2] = { bias[tn + nlo], bias[tn + 16 + nlo] };
#pragma unroll
  for (int s = 0; s < 2; ++s)
#pragma unroll
    for (int ns = 0; ns < 2; ++ns)
#pragma unroll
      for (int v = 0; v < 8; ++v) {
        float x = acc[s][ns][v] + bv[ns];
        if (act == 1) x = 0.5f * x * (1.0f + erff(x * 0.70710678118f));
        size_t row = (size_t)tm + s * 16 + hi8 + v;
        size_t col = (size_t)tn + ns * 16 + nlo;
        if (outF) outF[row * N + col] = x;
        if (outH) outH[row * N + col] = (_Float16)x;
      }
}

// ---------------------------------------------------------------------------
// Stage E: window reverse + roll back + LayerNorm(norm1) + residual.
// One wave per token; writes x1 in f32 (residual path) and f16 (fc1 input).
// ---------------------------------------------------------------------------
__global__ void fuse_ln1_kernel(const float* __restrict__ yproj, const float* __restrict__ x,
                                const float* __restrict__ n1w, const float* __restrict__ n1b,
                                float* __restrict__ x1f, _Float16* __restrict__ x1h) {
  int wave = threadIdx.x >> 5, lane = threadIdx.x & 31;
  int T = blockIdx.x * 8 + wave;                     // b*4096 + gr*64 + gc
  int b = T >> 12, gr = (T >> 6) & 63, gc = T & 63;
  int sr = (gr + HRES - SS) & 63, sc = (gc + WRES - SS) & 63;
  int w = (b << 6) + ((sr >> 3) << 3) + (sc >> 3);
  int t = ((sr & 7) << 3) + (sc & 7);
  const float* src = yproj + (size_t)(w * NTOK + t) * DIMC;

  float vals[8];
  float mu = 0.0f;
#pragma unroll
  for (int i = 0; i < 8; ++i) { vals[i] = src[lane * 8 + i]; mu += vals[i]; }
#pragma unroll
  for (int m = 16; m >= 1; m >>= 1) mu += __shfl_xor(mu, m, 32);
  mu *= (1.0f / (float)DIMC);
  float var = 0.0f;
#pragma unroll
  for (int i = 0; i < 8; ++i) { float d = vals[i] - mu; var += d * d; }
#pragma unroll
  for (int m = 16; m >= 1; m >>= 1) var += __shfl_xor(var, m, 32);
  var *= (1.0f / (float)DIMC);
  float inv = rsqrtf(var + EPS_LN);
#pragma unroll
  for (int i = 0; i < 8; ++i) {
    int c = lane * 8 + i;
    float o = (vals[i] - mu) * inv * n1w[c] + n1b[c] + x[(size_t)T * DIMC + c];
    x1f[(size_t)T * DIMC + c] = o;
    x1h[(size_t)T * DIMC + c] = (_Float16)o;
  }
}

// ---------------------------------------------------------------------------
// Stage H: out = x1 + LayerNorm(g, norm2). One wave per token.
// ---------------------------------------------------------------------------
__global__ void fuse_ln2_kernel(const float* __restrict__ g, const float* __restrict__ x1f,
                                const float* __restrict__ n2w, const float* __restrict__ n2b,
                                float* __restrict__ out) {
  int wave = threadIdx.x >> 5, lane = threadIdx.x & 31;
  size_t T = (size_t)blockIdx.x * 8 + wave;
  const float* src = g + T * DIMC;
  float vals[8];
  float mu = 0.0f;
#pragma unroll
  for (int i = 0; i < 8; ++i) { vals[i] = src[lane * 8 + i]; mu += vals[i]; }
#pragma unroll
  for (int m = 16; m >= 1; m >>= 1) mu += __shfl_xor(mu, m, 32);
  mu *= (1.0f / (float)DIMC);
  float var = 0.0f;
#pragma unroll
  for (int i = 0; i < 8; ++i) { float d = vals[i] - mu; var += d * d; }
#pragma unroll
  for (int m = 16; m >= 1; m >>= 1) var += __shfl_xor(var, m, 32);
  var *= (1.0f / (float)DIMC);
  float inv = rsqrtf(var + EPS_LN);
#pragma unroll
  for (int i = 0; i < 8; ++i) {
    int c = lane * 8 + i;
    out[T * DIMC + c] = x1f[T * DIMC + c] + (vals[i] - mu) * inv * n2w[c] + n2b[c];
  }
}

// ---------------------------------------------------------------------------
// Launch
// ---------------------------------------------------------------------------
extern "C" void kernel_launch(void* const* d_in, const int* in_sizes, int n_in,
                              void* d_out, int out_size, void* d_ws, size_t ws_size,
                              hipStream_t stream) {
  const float* x      = (const float*)d_in[0];
  const float* qkv_w  = (const float*)d_in[1];
  const float* qkv_b  = (const float*)d_in[2];
  const float* nq_w   = (const float*)d_in[3];
  const float* nk_w   = (const float*)d_in[4];
  const float* cpb_w1 = (const float*)d_in[5];
  const float* cpb_b1 = (const float*)d_in[6];
  const float* cpb_w2 = (const float*)d_in[7];
  const float* proj_w = (const float*)d_in[8];
  const float* proj_b = (const float*)d_in[9];
  const float* n1_w   = (const float*)d_in[10];
  const float* n1_b   = (const float*)d_in[11];
  const float* fc1_w  = (const float*)d_in[12];
  const float* fc1_b  = (const float*)d_in[13];
  const float* fc2_w  = (const float*)d_in[14];
  const float* fc2_b  = (const float*)d_in[15];
  const float* n2_w   = (const float*)d_in[16];
  const float* n2_b   = (const float*)d_in[17];
  float* out = (float*)d_out;

  char* ws = (char*)d_ws;
  _Float16* qkvw  = (_Float16*)(ws + OFF_QKVW);
  _Float16* projw = (_Float16*)(ws + OFF_PROJW);
  _Float16* fc1w  = (_Float16*)(ws + OFF_FC1W);
  _Float16* fc2w  = (_Float16*)(ws + OFF_FC2W);
  float*    tab   = (float*)(ws + OFF_TAB);
  float*    bias  = (float*)(ws + OFF_BIAS);
  _Float16* xw    = (_Float16*)(ws + OFF_XW);
  _Float16* qb    = (_Float16*)(ws + OFF_Q);
  _Float16* kb    = (_Float16*)(ws + OFF_K);
  _Float16* vtb   = (_Float16*)(ws + OFF_VT);
  _Float16* y16   = (_Float16*)(ws + OFF_Y16);
  float*    yproj = (float*)(ws + OFF_YPROJ);
  _Float16* h16   = (_Float16*)(ws + OFF_H16);
  float*    x1f   = (float*)(ws + OFF_X1F32);
  _Float16* x1h   = (_Float16*)(ws + OFF_X1F16);
  float*    gbuf  = (float*)(ws + OFF_G);

  // weight conversion
  cvt_f16_kernel<<<768, 256, 0, stream>>>(qkv_w, qkvw, 3 * DIMC * DIMC);
  cvt_f16_kernel<<<256, 256, 0, stream>>>(proj_w, projw, DIMC * DIMC);
  cvt_f16_kernel<<<1024, 256, 0, stream>>>(fc1_w, fc1w, 4 * DIMC * DIMC);
  cvt_f16_kernel<<<1024, 256, 0, stream>>>(fc2_w, fc2w, 4 * DIMC * DIMC);

  // CPB relative position bias
  cpb_tab_kernel<<<225, 256, 0, stream>>>(cpb_w1, cpb_b1, cpb_w2, tab);
  cpb_bias_kernel<<<128, 256, 0, stream>>>(tab, bias);

  // shifted window partition (f16)
  winpart_kernel<<<MROWS, 256, 0, stream>>>(x, xw);

  // QKV GEMM + RMSNorm epilogue: tiles = 1024 * 24 = 24576 -> 3072 blocks
  qkv_gemm_kernel<<<3072, 256, 0, stream>>>(xw, qkvw, qkv_b, nq_w, nk_w, qb, kb, vtb);

  // attention per (window, head): 512*8 = 4096 workgroups of 4 waves
  attn_kernel<<<4096, 128, 0, stream>>>(qb, kb, vtb, bias, y16);

  // proj GEMM: tiles = 1024*8 -> 1024 blocks
  gemm_wmma_kernel<<<1024, 256, 0, stream>>>(y16, projw, proj_b, yproj, (_Float16*)nullptr,
                                             DIMC, DIMC, 0);

  // window reverse + LN1 + residual
  fuse_ln1_kernel<<<MROWS / 8, 256, 0, stream>>>(yproj, x, n1_w, n1_b, x1f, x1h);

  // fc1 + exact GELU (f16 out): tiles = 1024*32 -> 4096 blocks
  gemm_wmma_kernel<<<4096, 256, 0, stream>>>(x1h, fc1w, fc1_b, (float*)nullptr, h16,
                                             4 * DIMC, DIMC, 1);

  // fc2 (f32 out): tiles = 1024*8 -> 1024 blocks, K = 1024
  gemm_wmma_kernel<<<1024, 256, 0, stream>>>(h16, fc2w, fc2_b, gbuf, (_Float16*)nullptr,
                                             DIMC, 4 * DIMC, 0);

  // final: out = x1 + LN2(g)
  fuse_ln2_kernel<<<MROWS / 8, 256, 0, stream>>>(gbuf, x1f, n2_w, n2_b, out);
}
